// LSTM_91027536872108
// MI455X (gfx1250) — matrix operand order repaired and checked by reading, not compiled, not run on hardware
//
#include <hip/hip_runtime.h>
#include <hip/hip_bf16.h>

// ---------------------------------------------------------------------------
// LSTM on gfx1250: per-timestep fused GEMM (f16-split hi/lo -> ~fp32 accuracy)
// v_wmma_f32_16x16x32_f16 core, weight fragments staged via async global->LDS
// (ASYNCcnt), K split across wave pairs with LDS reduction, fused epilogue.
// ---------------------------------------------------------------------------

#define T_STEPS 512
#define BATCH   256
#define IN_DIM  64
#define AUX_DIM 64
#define OUT_DIM 512
#define GATE_IN 640            // 128 x-part + 512 h-part
#define NKC     20             // K chunks of 32: 640/32
#define KHALF   10             // chunks per K-half
#define NTILES_N 32            // OUT_DIM/16
#define NTILES_M 16            // BATCH/16

typedef __attribute__((ext_vector_type(16))) _Float16 v16h;
typedef __attribute__((ext_vector_type(8)))  float    v8f;
typedef __attribute__((ext_vector_type(4)))  float    f32x4;

// workspace layout (in _Float16 units)
#define WPH_OFF   0u                         // packed W hi: 4*32*20*512 = 1,310,720 halfs
#define WPL_OFF   1310720u                   // packed W lo
#define A0H_OFF   2621440u                   // A plane buf0 hi: 256*640 = 163,840 halfs
#define A0L_OFF   2785280u
#define A1H_OFF   2949120u
#define A1L_OFF   3112960u
// total: 3,276,800 halfs = 6,553,600 bytes

// LDS: 3 staging buffers x 16 fragments x 1KB = 48KB (reduction area reuses buf0)
#define LDS_BUF_BYTES  16384
#define LDS_TOTAL      49152

__device__ __forceinline__ void ld16h(v16h& dst, const _Float16* p0, const _Float16* p1) {
    ((f32x4*)&dst)[0] = *(const f32x4*)p0;   // 8 halfs
    ((f32x4*)&dst)[1] = *(const f32x4*)p1;   // 8 halfs
}

__device__ __forceinline__ float sigmoidf_(float x) { return 1.0f / (1.0f + expf(-x)); }

// async copy of two 16B-per-lane slices (offsets 0 and 512), global(saddr+voff) -> LDS(ldsaddr)
__device__ __forceinline__ void async_b128(unsigned ldsaddr, unsigned voff,
                                           unsigned long long saddr) {
    asm volatile("global_load_async_to_lds_b128 %0, %1, %2 offset:0"
                 :: "v"(ldsaddr), "v"(voff), "s"(saddr) : "memory");
    asm volatile("global_load_async_to_lds_b128 %0, %1, %2 offset:512"
                 :: "v"(ldsaddr), "v"(voff), "s"(saddr) : "memory");
}

// ---------------------------------------------------------------------------
// Pack the 4 gate weight matrices [640,512] fp32 into f16 hi/lo planes in a
// fragment-major layout: frag(g, tn, kc) is 512 contiguous halfs, lane L owns
// halfs [L*16 .. L*16+15] = (K = kc*32 + 16*(L/16) + j, N = tn*16 + L%16).
// ---------------------------------------------------------------------------
__global__ __launch_bounds__(256) void lstm_pack_weights(
    const float* __restrict__ Wf, const float* __restrict__ Wi,
    const float* __restrict__ Wo, const float* __restrict__ Wp,
    _Float16* __restrict__ wph, _Float16* __restrict__ wpl)
{
    int idx = blockIdx.x * 256 + threadIdx.x;          // 4*640*512 = 1,310,720
    if (idx >= 4 * GATE_IN * OUT_DIM) return;
    int g   = idx / (GATE_IN * OUT_DIM);
    int rem = idx % (GATE_IN * OUT_DIM);
    int k   = rem / OUT_DIM;
    int n   = rem % OUT_DIM;
    const float* W = (g == 0) ? Wf : (g == 1) ? Wi : (g == 2) ? Wo : Wp;
    float v = W[k * OUT_DIM + n];
    _Float16 hi = (_Float16)v;
    _Float16 lo = (_Float16)(v - (float)hi);
    int tn = n >> 4, nn = n & 15;
    int kc = k >> 5, kk = k & 31;
    int L  = ((kk >> 4) << 4) + nn;                    // lane
    int j  = kk & 15;
    unsigned off = (((unsigned)(g * NTILES_N + tn) * NKC + kc) * 512u) + L * 16 + j;
    wph[off] = hi;
    wpl[off] = lo;
}

// ---------------------------------------------------------------------------
// Build A-plane (z-matrix, [256 x 640], f16 hi/lo) for t = 0.
// ---------------------------------------------------------------------------
__global__ __launch_bounds__(256) void lstm_init_A(
    const float* __restrict__ xm, const float* __restrict__ xa,
    _Float16* __restrict__ ah, _Float16* __restrict__ al)
{
    int idx = blockIdx.x * 256 + threadIdx.x;          // 256*640 = 163,840
    if (idx >= BATCH * GATE_IN) return;
    int b = idx / GATE_IN;
    int k = idx % GATE_IN;
    float v = 0.0f;
    if (k < IN_DIM)            v = xm[b * IN_DIM + k];
    else if (k < 128)          v = xa[b * AUX_DIM + (k - IN_DIM)];
    _Float16 hi = (_Float16)v;
    ah[idx] = hi;
    al[idx] = (_Float16)(v - (float)hi);
}

// ---------------------------------------------------------------------------
// One LSTM timestep.
// Grid: 128 blocks = (nt in 0..31) x (mgroup in 0..3); block = 8 waves:
//   wave w: mt = mgroup*4 + (w&3), khalf = w>>2 (K 0..319 or 320..639).
// All 8 waves share one n-tile -> weight fragments staged once per block into
// LDS via async global->LDS loads (triple buffered, ASYNCcnt + barrier sync).
// K-halves reduced through LDS; low-half waves run the fused LSTM epilogue.
// ---------------------------------------------------------------------------
__global__ __launch_bounds__(256) void lstm_step(
    int t,
    const _Float16* __restrict__ ahi_in,  const _Float16* __restrict__ alo_in,
    _Float16* __restrict__ ahi_out,       _Float16* __restrict__ alo_out,
    const _Float16* __restrict__ wph,     const _Float16* __restrict__ wpl,
    const float* __restrict__ bf, const float* __restrict__ bi,
    const float* __restrict__ bo, const float* __restrict__ bp,
    const float* __restrict__ xm, const float* __restrict__ xa,
    float* __restrict__ hs, float* __restrict__ cs)
{
    __shared__ __align__(16) unsigned char smem[LDS_TOTAL];
    const unsigned sbase = (unsigned)(size_t)(void*)smem;   // LDS byte offset

    const int lane  = threadIdx.x & 31;
    const int w     = threadIdx.x >> 5;        // 0..7
    const int nt    = blockIdx.x >> 2;         // 0..31
    const int mgrp  = blockIdx.x & 3;          // 0..3
    const int mt    = mgrp * 4 + (w & 3);      // 0..15
    const int kh    = w >> 2;                  // K-half: 0 or 1
    const int m0    = mt << 4;
    const int n0    = nt << 4;
    const int g2    = lane >> 4;
    const int l16   = lane & 15;

    // --- async staging: wave w issues fragments {2w, 2w+1} of 16 per chunk ---
    // fragment f: fkh = f>>3, gate = (f>>1)&3, plane = f&1
    const int f0   = 2 * w;
    const unsigned long long wphA = (unsigned long long)(size_t)wph;
    const unsigned long long wplA = (unsigned long long)(size_t)wpl;

    auto issue = [&](int kcc, int buf) {
#pragma unroll
        for (int fi = 0; fi < 2; ++fi) {
            int f     = f0 + fi;
            int fkh   = f >> 3;
            int gate  = (f >> 1) & 3;
            int plane = f & 1;
            int kc    = fkh * KHALF + kcc;
            unsigned gOff = ((((unsigned)(gate * NTILES_N + nt) * NKC + kc) * 512u) << 1)
                            + lane * 16;                       // bytes
            unsigned lOff = sbase + buf * LDS_BUF_BYTES + f * 1024 + lane * 16;
            async_b128(lOff, gOff, plane ? wplA : wphA);
        }
    };

    v8f acc[4] = {};

    const _Float16* aRowH = ahi_in + (m0 + l16) * GATE_IN;
    const _Float16* aRowL = alo_in + (m0 + l16) * GATE_IN;

    issue(0, 0);   // prologue

    for (int kcc = 0; kcc < KHALF; ++kcc) {
        if (kcc + 1 < KHALF) {
            issue(kcc + 1, (kcc + 1) % 3);
            asm volatile("s_wait_asynccnt 0x4" ::: "memory");
        } else {
            asm volatile("s_wait_asynccnt 0x0" ::: "memory");
        }
        __syncthreads();   // all waves' fragments for kcc are in LDS

        const int k0 = (kh * KHALF + kcc) << 5;
        v16h ah, al;
        ld16h(ah, aRowH + k0 + 16 * g2, aRowH + k0 + 16 * g2 + 8);
        ld16h(al, aRowL + k0 + 16 * g2, aRowL + k0 + 16 * g2 + 8);

        const unsigned char* bufp = smem + (kcc % 3) * LDS_BUF_BYTES + kh * 8192;
#pragma unroll
        for (int g = 0; g < 4; ++g) {
            const f32x4* bhp = (const f32x4*)(bufp + (g * 2 + 0) * 1024 + lane * 32);
            const f32x4* blp = (const f32x4*)(bufp + (g * 2 + 1) * 1024 + lane * 32);
            v16h bh, bl;
            ((f32x4*)&bh)[0] = bhp[0];  ((f32x4*)&bh)[1] = bhp[1];
            ((f32x4*)&bl)[0] = blp[0];  ((f32x4*)&bl)[1] = blp[1];
            // f16-split product: a*b ~= ahi*bhi + ahi*blo + alo*bhi (fp32 accum)
            acc[g] = __builtin_amdgcn_wmma_f32_16x16x32_f16(false, ah, false, bh, (short)0, acc[g], false, false);
            acc[g] = __builtin_amdgcn_wmma_f32_16x16x32_f16(false, ah, false, bl, (short)0, acc[g], false, false);
            acc[g] = __builtin_amdgcn_wmma_f32_16x16x32_f16(false, al, false, bh, (short)0, acc[g], false, false);
        }
    }

    // --- K-half reduction through LDS (reuse staging area) ---
    __syncthreads();                     // everyone done reading staging bufs
    if (kh == 1) {
        unsigned char* red = smem + (w - 4) * 4096 + lane * 128;
#pragma unroll
        for (int g = 0; g < 4; ++g) {
            ((f32x4*)(red + g * 32))[0] = ((f32x4*)&acc[g])[0];
            ((f32x4*)(red + g * 32))[1] = ((f32x4*)&acc[g])[1];
        }
    }
    __syncthreads();

    if (kh == 0) {
        const unsigned char* red = smem + (w & 3) * 4096 + lane * 128;
#pragma unroll
        for (int g = 0; g < 4; ++g) {
            f32x4 p0 = ((const f32x4*)(red + g * 32))[0];
            f32x4 p1 = ((const f32x4*)(red + g * 32))[1];
            ((f32x4*)&acc[g])[0] += p0;
            ((f32x4*)&acc[g])[1] += p1;
        }

        // ---- fused LSTM epilogue ----
        // C/D layout: elem r -> row M = m0 + r + 8*(lane/16), col N = n0 + lane%16.
        const int   col = n0 + l16;
        const float bF = bf[col], bI = bi[col], bO = bo[col], bP = bp[col];
        float*       hs_t = hs + (size_t)t * BATCH * OUT_DIM;
        float*       cs_t = cs + (size_t)t * BATCH * OUT_DIM;
        const float* cs_p = cs + (size_t)(t - 1) * BATCH * OUT_DIM;  // read iff t>0

#pragma unroll
        for (int r = 0; r < 8; ++r) {
            int row = m0 + r + 8 * g2;
            size_t off = (size_t)row * OUT_DIM + col;
            float cp = (t > 0) ? cs_p[off] : 0.0f;
            float fv = sigmoidf_(acc[0][r] + bF);
            float iv = sigmoidf_(acc[1][r] + bI);
            float ov = sigmoidf_(acc[2][r] + bO);
            float pv = tanhf(acc[3][r] + bP);
            float cn = fv * cp + iv * pv;
            float hn = ov * tanhf(cn);
            cs_t[off] = cn;
            hs_t[off] = hn;
            _Float16 hh = (_Float16)hn;                 // split h for next step
            int aoff = row * GATE_IN + 128 + col;
            ahi_out[aoff] = hh;
            alo_out[aoff] = (_Float16)(hn - (float)hh);
        }
    }

    // ---- prepare x-part of A-plane for step t+1 (32768 threads, 1 elem each) ----
    if (t + 1 < T_STEPS) {
        const float* xmT = xm + (size_t)(t + 1) * BATCH * IN_DIM;
        const float* xaT = xa + (size_t)(t + 1) * BATCH * AUX_DIM;
        int i = blockIdx.x * 256 + threadIdx.x;        // 0..32767 = 256*128
        int b = i >> 7;
        int k = i & 127;
        float v = (k < IN_DIM) ? xmT[b * IN_DIM + k] : xaT[b * AUX_DIM + (k - IN_DIM)];
        _Float16 vh = (_Float16)v;
        int aoff = b * GATE_IN + k;
        ahi_out[aoff] = vh;
        alo_out[aoff] = (_Float16)(v - (float)vh);
    }
}

// ---------------------------------------------------------------------------
extern "C" void kernel_launch(void* const* d_in, const int* in_sizes, int n_in,
                              void* d_out, int out_size, void* d_ws, size_t ws_size,
                              hipStream_t stream) {
    const float* xm = (const float*)d_in[0];
    const float* xa = (const float*)d_in[1];
    const float* Wf = (const float*)d_in[2];
    const float* bf = (const float*)d_in[3];
    const float* Wi = (const float*)d_in[4];
    const float* bi = (const float*)d_in[5];
    const float* Wo = (const float*)d_in[6];
    const float* bo = (const float*)d_in[7];
    const float* Wp = (const float*)d_in[8];
    const float* bp = (const float*)d_in[9];

    float* hs = (float*)d_out;                                   // [T,B,512]
    float* cs = hs + (size_t)T_STEPS * BATCH * OUT_DIM;          // [T,B,512]

    _Float16* wsh = (_Float16*)d_ws;
    _Float16* wph = wsh + WPH_OFF;
    _Float16* wpl = wsh + WPL_OFF;
    _Float16* aH[2] = { wsh + A0H_OFF, wsh + A1H_OFF };
    _Float16* aL[2] = { wsh + A0L_OFF, wsh + A1L_OFF };

    // 1) repack weights into WMMA fragment layout (f16 hi/lo)
    lstm_pack_weights<<<(4 * GATE_IN * OUT_DIM + 255) / 256, 256, 0, stream>>>(
        Wf, Wi, Wo, Wp, wph, wpl);

    // 2) build A-plane for t = 0 (x part + zero h0)
    lstm_init_A<<<(BATCH * GATE_IN + 255) / 256, 256, 0, stream>>>(xm, xa, aH[0], aL[0]);

    // 3) recurrent loop: one fused GEMM+epilogue kernel per timestep
    for (int t = 0; t < T_STEPS; ++t) {
        int in = t & 1, out = (t + 1) & 1;
        lstm_step<<<128, 256, 0, stream>>>(
            t, aH[in], aL[in], aH[out], aL[out], wph, wpl,
            bf, bi, bo, bp, xm, xa, hs, cs);
    }
}
